// Decoder_47261820125459
// MI455X (gfx1250) — matrix operand order repaired and checked by reading, not compiled, not run on hardware
//
#include <hip/hip_runtime.h>
#include <math.h>
#include <stdint.h>

// ---------------- types ----------------
typedef __bf16 bf16_t;
typedef __attribute__((ext_vector_type(8)))  __bf16 v8bf;
typedef __attribute__((ext_vector_type(16))) __bf16 v16bf;
typedef __attribute__((ext_vector_type(8)))  float  v8f;

// ---------------- problem constants ----------------
#define B_    4
#define S_    1024
#define K_    1024
#define D_    1024
#define H_    16
#define HD_   64
#define COND_ 1024
#define MR_   4
__device__ constexpr float INV_SQRT_HD = 0.125f;          // 1/sqrt(64)
__device__ constexpr float GELU_CONST  = 0.7978845608028654f;

// epilogue kinds
constexpr int EPI_ROPE_Q = 0;   // q = (qn@Wq) -> rope -> bf16
constexpr int EPI_KV     = 1;   // kv = (kvn@Wkv): k -> rope -> bf16 rowmajor, v -> bf16 transposed per head
constexpr int EPI_WO     = 2;   // x = gate ? g_msa*(attn@Wo)+q_x : q_x  (fp32)
constexpr int EPI_GELU   = 3;   // h1 = gelu(x@W1 + b1) -> bf16
constexpr int EPI_FINAL  = 4;   // out = gate ? g_mlp*(h1@W2+b2)+x_mid : x_mid (fp32)

// ---------------- helpers ----------------
__device__ inline v16bf concat8(v8bf lo, v8bf hi) {
  return __builtin_shufflevector(lo, hi, 0,1,2,3,4,5,6,7,8,9,10,11,12,13,14,15);
}

__device__ inline v8f wmma_bf16(v16bf a, v16bf b, v8f c) {
  // D = A(16x32) * B(32x16) + C, fp32 accum
  return __builtin_amdgcn_wmma_f32_16x16x32_bf16(false, a, false, b, (short)0, c, false, false);
}

// A fragment (contiguous k order, same as the B fragment):
//  lanes 0-15 : row = m0+lane,    K = kk..kk+15
//  lanes 16-31: row = m0+lane-16, K = kk+16..kk+31
__device__ inline v16bf load_a_frag(const bf16_t* __restrict__ A, int rowStride,
                                    int m0, int kk, int lane) {
  int row  = m0 + (lane & 15);
  int base = kk + ((lane >> 4) << 4);
  const bf16_t* p = A + (size_t)row * rowStride + base;
  v8bf lo = *(const v8bf*)p;
  v8bf hi = *(const v8bf*)(p + 8);
  return concat8(lo, hi);
}

// ---------------- adaLN modulation: mod = t_cond @ adaW + adab ----------------
__global__ void mod_kernel(const float* __restrict__ t_cond, const float* __restrict__ adaW,
                           const float* __restrict__ adab, float* __restrict__ mod) {
  int j = blockIdx.x * blockDim.x + threadIdx.x;   // 0 .. 6D-1
  int b = blockIdx.y;
  const float* t = t_cond + (size_t)b * COND_;
  float acc = 0.f;
  for (int c = 0; c < COND_; ++c) acc = fmaf(t[c], adaW[(size_t)c * (6 * D_) + j], acc);
  mod[(size_t)b * 6 * D_ + j] = acc + adab[j];
}

// ---------------- weight transpose fp32 -> bf16 : dst[c][r] = src[r][c] ----------------
__global__ void transpose_to_bf16(const float* __restrict__ src, bf16_t* __restrict__ dst,
                                  int R, int C) {
  __shared__ float tile[32][33];
  int c0 = blockIdx.x * 32, r0 = blockIdx.y * 32;
  int tx = threadIdx.x & 31, ty = threadIdx.x >> 5;   // 256 threads: ty 0..7
  for (int i = ty; i < 32; i += 8)
    tile[i][tx] = src[(size_t)(r0 + i) * C + c0 + tx];
  __syncthreads();
  for (int i = ty; i < 32; i += 8)
    dst[(size_t)(c0 + i) * R + r0 + tx] = (bf16_t)tile[tx][i];
}

// ---------------- LayerNorm (+ optional adaLN mod) -> bf16 ----------------
// MODE 0: q path  (w=qn_w,  *(1+sc_msa)+sh_msa)
// MODE 1: kv path (w=kvn_w, no mod)
// MODE 2: mlp path(w=n2_w,  *(1+sc_mlp)+sh_mlp)
template<int MODE>
__global__ void ln_kernel(const float* __restrict__ x, const float* __restrict__ w,
                          const float* __restrict__ mod, bf16_t* __restrict__ out,
                          int rowsPerBatch) {
  __shared__ float red[32];
  int row = blockIdx.x;
  int b   = row / rowsPerBatch;
  const float* xr = x + (size_t)row * D_;
  int t = threadIdx.x;
  float v[4], s = 0.f, s2 = 0.f;
  #pragma unroll
  for (int i = 0; i < 4; ++i) { v[i] = xr[t + i * 256]; s += v[i]; s2 += v[i] * v[i]; }
  #pragma unroll
  for (int m = 16; m >= 1; m >>= 1) { s += __shfl_xor(s, m, 32); s2 += __shfl_xor(s2, m, 32); }
  if ((t & 31) == 0) { red[t >> 5] = s; red[8 + (t >> 5)] = s2; }
  __syncthreads();
  if (t == 0) {
    float a = 0.f, c = 0.f;
    for (int i = 0; i < 8; ++i) { a += red[i]; c += red[8 + i]; }
    red[16] = a; red[17] = c;
  }
  __syncthreads();
  float mu  = red[16] * (1.0f / D_);
  float var = red[17] * (1.0f / D_) - mu * mu;
  float inv = rsqrtf(var + 1e-5f);
  #pragma unroll
  for (int i = 0; i < 4; ++i) {
    int c = t + i * 256;
    float y = (v[i] - mu) * inv * w[c];
    if (MODE == 0) y = y * (1.f + mod[(size_t)b * 6 * D_ + D_ + c])     + mod[(size_t)b * 6 * D_ + c];
    if (MODE == 2) y = y * (1.f + mod[(size_t)b * 6 * D_ + 4 * D_ + c]) + mod[(size_t)b * 6 * D_ + 3 * D_ + c];
    out[(size_t)row * D_ + c] = (bf16_t)y;
  }
}

// ---------------- WMMA GEMM: 32Mx64N per wave, 256Mx64N per block ----------------
// C(M,N) = A(M,Kd) @ Bt(N,Kd)^T.
// All 8 waves of a block share one 64(N) x 32(K) B tile, double-buffered in LDS and
// filled with gfx1250 async copies (global_load_async_to_lds_b128, ASYNCcnt-tracked).
template<int EPI>
__global__ void __launch_bounds__(256)
gemm_wmma(const bf16_t* __restrict__ A, const bf16_t* __restrict__ Bt,
          int M, int N, int Kd,
          const float* __restrict__ e0,   // cos | q_x | b1 | b2
          const float* __restrict__ e1,   // sin | xF(mid)
          const unsigned char* __restrict__ gate,
          const float* __restrict__ mod,
          bf16_t* __restrict__ outB, float* __restrict__ outF,
          bf16_t* __restrict__ outB2) {
  __shared__ __align__(16) bf16_t ldsB[2][64][32];    // 8 KB double buffer
  int lane = threadIdx.x & 31;
  int wl   = threadIdx.x >> 5;
  int mBlocks = M >> 8;                               // 256 rows / block
  int mBlk = blockIdx.x % mBlocks;
  int nT   = blockIdx.x / mBlocks;                    // consecutive blocks share B strip
  int m0 = (mBlk << 8) + (wl << 5);                   // wave's 32 rows
  int n0 = nT << 6;

  // cooperative B-tile staging: thread t copies 16 bytes (64 cols x 32 K x bf16 = 4 KB)
  int scol   = threadIdx.x >> 2;                      // 0..63
  int schunk = (threadIdx.x & 3) << 3;                // element offset 0/8/16/24
  auto stage = [&](int buf, int kk) {
    unsigned dst = (unsigned)(uintptr_t)&ldsB[buf][scol][schunk];
    const bf16_t* src = Bt + (size_t)(n0 + scol) * Kd + kk + schunk;
    asm volatile("global_load_async_to_lds_b128 %0, %1, off"
                 :: "v"(dst), "v"((unsigned long long)(uintptr_t)src)
                 : "memory");
  };

  v8f acc[2][4] = {};
  stage(0, 0);
  int buf = 0;
  for (int kk = 0; kk < Kd; kk += 32) {
    asm volatile("s_wait_asynccnt 0x0" ::: "memory"); // own async stores landed in LDS
    __syncthreads();                                  // whole tile visible to all waves
    if (kk + 32 < Kd) stage(buf ^ 1, kk + 32);        // prefetch next K-slab
    __builtin_prefetch(A + (size_t)(m0 + (lane & 15)) * Kd + kk + 128, 0, 3);
    v16bf a0 = load_a_frag(A, Kd, m0,      kk, lane);
    v16bf a1 = load_a_frag(A, Kd, m0 + 16, kk, lane);
    #pragma unroll
    for (int j = 0; j < 4; ++j) {
      // B fragment from LDS: lane<16 -> col n0+j*16+lane, K kk+0..15; lane>=16 -> K kk+16..31
      const bf16_t* bp = &ldsB[buf][j * 16 + (lane & 15)][(lane >> 4) << 4];
      v16bf b = *(const v16bf*)bp;
      acc[0][j] = wmma_bf16(a0, b, acc[0][j]);
      acc[1][j] = wmma_bf16(a1, b, acc[1][j]);
    }
    __syncthreads();                                  // reads done before this buf is restaged
    buf ^= 1;
  }

  int hi8 = (lane >> 4) << 3;
  int lc  = lane & 15;
  #pragma unroll
  for (int mt = 0; mt < 2; ++mt) {
    #pragma unroll
    for (int j = 0; j < 4; ++j) {
      #pragma unroll
      for (int r = 0; r < 8; ++r) {
        int m = m0 + mt * 16 + r + hi8;
        int n = n0 + j * 16 + lc;
        float c = acc[mt][j][r];
        if constexpr (EPI == EPI_ROPE_Q) {
          int s = m & (S_ - 1);
          int d = n & (HD_ - 1);
          float other = acc[mt][j ^ 2][r];            // partner column (+/-32 within head)
          float rot = (j < 2) ? -other : other;
          float y = c * e0[s * HD_ + d] + rot * e1[s * HD_ + d];
          outB[(size_t)m * D_ + n] = (bf16_t)y;
        } else if constexpr (EPI == EPI_KV) {
          int s  = m & (K_ - 1);
          int bb = m >> 10;                           // m / K_
          if (n < D_) {                               // K projection: rope, row-major
            int d = n & (HD_ - 1);
            float other = acc[mt][j ^ 2][r];
            float rot = (j < 2) ? -other : other;
            float y = c * e0[s * HD_ + d] + rot * e1[s * HD_ + d];
            outB[(size_t)m * D_ + n] = (bf16_t)y;
          } else {                                    // V projection: store transposed per head
            int nn = n - D_;
            int h = nn >> 6, d = nn & (HD_ - 1);
            outB2[(((size_t)bb * H_ + h) * HD_ + d) * K_ + s] = (bf16_t)c;
          }
        } else if constexpr (EPI == EPI_WO) {
          int bb = m / S_;
          float res = e0[(size_t)m * D_ + n];         // q_x
          float y = mod[(size_t)bb * 6 * D_ + 2 * D_ + n] * c + res;
          outF[(size_t)m * D_ + n] = gate[m] ? y : res;
        } else if constexpr (EPI == EPI_GELU) {
          float y = c + e0[n];                        // + b1
          float u = GELU_CONST * (y + 0.044715f * y * y * y);
          outB[(size_t)m * N + n] = (bf16_t)(0.5f * y * (1.f + tanhf(u)));
        } else {                                      // EPI_FINAL
          int bb = m / S_;
          float y  = c + e0[n];                       // + b2
          float xm = e1[(size_t)m * D_ + n];          // x_mid
          float o  = mod[(size_t)bb * 6 * D_ + 5 * D_ + n] * y + xm;
          outF[(size_t)m * D_ + n] = gate[m] ? o : xm;
        }
      }
    }
  }
}

// ---------------- flash attention: one wave = (b, h, 16 query rows) ----------------
__global__ void attn_kernel(const bf16_t* __restrict__ qB, const bf16_t* __restrict__ kB,
                            const bf16_t* __restrict__ vT, const int* __restrict__ gidx,
                            bf16_t* __restrict__ attnB) {
  __shared__ __bf16 ldsP[8][16][32];                  // per-wave P tile (C-layout -> A-layout bounce)
  int lane = threadIdx.x & 31;
  int wl   = threadIdx.x >> 5;
  int w    = blockIdx.x * 8 + wl;
  int qt = w & 63;                                    // S/16 tiles
  int h  = (w >> 6) & (H_ - 1);
  int b  = w >> 10;
  int s0 = qt << 4;
  int hi = lane >> 4, lc = lane & 15;

  // Q fragments for hd 0..31 and 32..63
  v16bf aq[2];
  {
    const bf16_t* qp = qB + (size_t)(b * S_ + s0 + lc) * D_ + h * HD_;
    #pragma unroll
    for (int ks = 0; ks < 2; ++ks) {
      int base = ks * 32 + hi * 16;
      aq[ks] = concat8(*(const v8bf*)(qp + base), *(const v8bf*)(qp + base + 8));
    }
  }
  int gq[8];
  #pragma unroll
  for (int r = 0; r < 8; ++r) gq[r] = gidx[b * S_ + s0 + r + hi * 8];

  float mrow[8], lrow[8];
  #pragma unroll
  for (int r = 0; r < 8; ++r) { mrow[r] = -1e30f; lrow[r] = 0.f; }
  v8f o[4] = {};

  for (int kc = 0; kc < K_; kc += 32) {
    // ---- scores S = Q K^T (2 key tiles x 2 k-steps) ----
    v8f sc[2] = {};
    #pragma unroll
    for (int nt = 0; nt < 2; ++nt) {
      const bf16_t* kp = kB + (size_t)(b * K_ + kc + nt * 16 + lc) * D_ + h * HD_;
      #pragma unroll
      for (int ks = 0; ks < 2; ++ks) {
        v16bf bk = *(const v16bf*)(kp + ks * 32 + hi * 16);
        sc[nt] = wmma_bf16(aq[ks], bk, sc[nt]);
      }
    }
    int gk0 = gidx[b * K_ + kc + lc];
    int gk1 = gidx[b * K_ + kc + 16 + lc];
    float sv[2][8];
    #pragma unroll
    for (int r = 0; r < 8; ++r) {
      float a0 = sc[0][r] * INV_SQRT_HD, a1 = sc[1][r] * INV_SQRT_HD;
      sv[0][r] = (gq[r] != gk0) ? a0 : -1e9f;
      sv[1][r] = (gq[r] != gk1) ? a1 : -1e9f;
    }
    // ---- online softmax update (16-lane reductions) ----
    float fac[8];
    #pragma unroll
    for (int r = 0; r < 8; ++r) {
      float t = fmaxf(sv[0][r], sv[1][r]);
      #pragma unroll
      for (int mk = 8; mk >= 1; mk >>= 1) t = fmaxf(t, __shfl_xor(t, mk, 16));
      float mn = fmaxf(mrow[r], t);
      fac[r] = expf(mrow[r] - mn);
      mrow[r] = mn;
    }
    float rs[8];
    #pragma unroll
    for (int r = 0; r < 8; ++r) rs[r] = 0.f;
    #pragma unroll
    for (int nt = 0; nt < 2; ++nt)
      #pragma unroll
      for (int r = 0; r < 8; ++r) {
        float p = expf(sv[nt][r] - mrow[r]);
        rs[r] += p;
        ldsP[wl][r + hi * 8][nt * 16 + lc] = (__bf16)p;
      }
    #pragma unroll
    for (int r = 0; r < 8; ++r) {
      float t = rs[r];
      #pragma unroll
      for (int mk = 8; mk >= 1; mk >>= 1) t += __shfl_xor(t, mk, 16);
      lrow[r] = lrow[r] * fac[r] + t;
    }
    #pragma unroll
    for (int j = 0; j < 4; ++j)
      #pragma unroll
      for (int r = 0; r < 8; ++r) o[j][r] = o[j][r] * fac[r];

    // ---- P (C-layout) -> A-fragment via LDS; same-wave DS ops are in-order ----
    asm volatile("s_wait_dscnt 0x0" ::: "memory");
    const __bf16* pp = &ldsP[wl][lc][hi * 16];
    v16bf pa = concat8(*(const v8bf*)pp, *(const v8bf*)(pp + 8));

    // ---- O += P @ V (V transposed per head: contiguous B fragments) ----
    #pragma unroll
    for (int j = 0; j < 4; ++j) {
      const bf16_t* vp = vT + (((size_t)b * H_ + h) * HD_ + (j * 16 + lc)) * K_ + kc + hi * 16;
      v16bf bv = *(const v16bf*)vp;
      o[j] = wmma_bf16(pa, bv, o[j]);
    }
  }

  #pragma unroll
  for (int r = 0; r < 8; ++r) {
    float inv = 1.f / fmaxf(lrow[r], 1e-20f);
    size_t row = (size_t)(b * S_ + s0 + r + hi * 8);
    #pragma unroll
    for (int j = 0; j < 4; ++j)
      attnB[row * D_ + h * HD_ + j * 16 + lc] = (bf16_t)(o[j][r] * inv);
  }
}

// ---------------- host orchestration ----------------
extern "C" void kernel_launch(void* const* d_in, const int* in_sizes, int n_in,
                              void* d_out, int out_size, void* d_ws, size_t ws_size,
                              hipStream_t stream) {
  (void)in_sizes; (void)n_in; (void)out_size; (void)ws_size;
  const float* q_x   = (const float*)d_in[0];
  const float* kv_x  = (const float*)d_in[1];
  const float* t_cond= (const float*)d_in[2];
  const float* cos_q = (const float*)d_in[3];
  const float* sin_q = (const float*)d_in[4];
  const float* cos_k = (const float*)d_in[5];
  const float* sin_k = (const float*)d_in[6];
  const int*   gidx  = (const int*)d_in[7];
  const unsigned char* qen = (const unsigned char*)d_in[8];   // bool8
  const float* qn_w  = (const float*)d_in[9];
  const float* kvn_w = (const float*)d_in[10];
  const float* n2_w  = (const float*)d_in[11];
  const float* Wq    = (const float*)d_in[12];
  const float* Wkv   = (const float*)d_in[13];
  const float* Wo    = (const float*)d_in[14];
  const float* W1    = (const float*)d_in[15];
  const float* b1    = (const float*)d_in[16];
  const float* W2    = (const float*)d_in[17];
  const float* b2    = (const float*)d_in[18];
  const float* adaW  = (const float*)d_in[19];
  const float* adab  = (const float*)d_in[20];

  char* ws = (char*)d_ws;
  size_t off = 0;
  auto alloc = [&](size_t bytes) -> void* {
    void* p = ws + off;
    off = (off + bytes + 255) & ~((size_t)255);
    return p;
  };
  float*  modF  = (float*) alloc((size_t)B_ * 6 * D_ * 4);
  bf16_t* WqT   = (bf16_t*)alloc((size_t)D_ * D_ * 2);
  bf16_t* WkvT  = (bf16_t*)alloc((size_t)2 * D_ * D_ * 2);
  bf16_t* WoT   = (bf16_t*)alloc((size_t)D_ * D_ * 2);
  bf16_t* W1T   = (bf16_t*)alloc((size_t)MR_ * D_ * D_ * 2);
  bf16_t* W2T   = (bf16_t*)alloc((size_t)MR_ * D_ * D_ * 2);
  bf16_t* qnB   = (bf16_t*)alloc((size_t)B_ * S_ * D_ * 2);   // reused for ln2 output
  bf16_t* kvnB  = (bf16_t*)alloc((size_t)B_ * K_ * D_ * 2);
  bf16_t* qBf   = (bf16_t*)alloc((size_t)B_ * S_ * D_ * 2);
  bf16_t* kBf   = (bf16_t*)alloc((size_t)B_ * K_ * D_ * 2);
  bf16_t* vTf   = (bf16_t*)alloc((size_t)B_ * K_ * D_ * 2);
  bf16_t* attnB = (bf16_t*)alloc((size_t)B_ * S_ * D_ * 2);
  float*  xF    = (float*) alloc((size_t)B_ * S_ * D_ * 4);
  bf16_t* h1B   = (bf16_t*)alloc((size_t)B_ * S_ * MR_ * D_ * 2);

  const int M = B_ * S_;                  // 4096 rows in every GEMM
  const int mBlocks = M / 256;            // 16

  // 1) adaLN modulation
  mod_kernel<<<dim3(6 * D_ / 256, B_), 256, 0, stream>>>(t_cond, adaW, adab, modF);
  // 2) weights -> transposed bf16 (L2-resident afterwards: ~24 MB << 192 MB)
  transpose_to_bf16<<<dim3(D_ / 32, D_ / 32), 256, 0, stream>>>(Wq, WqT, D_, D_);
  transpose_to_bf16<<<dim3(2 * D_ / 32, D_ / 32), 256, 0, stream>>>(Wkv, WkvT, D_, 2 * D_);
  transpose_to_bf16<<<dim3(D_ / 32, D_ / 32), 256, 0, stream>>>(Wo, WoT, D_, D_);
  transpose_to_bf16<<<dim3(MR_ * D_ / 32, D_ / 32), 256, 0, stream>>>(W1, W1T, D_, MR_ * D_);
  transpose_to_bf16<<<dim3(D_ / 32, MR_ * D_ / 32), 256, 0, stream>>>(W2, W2T, MR_ * D_, D_);
  // 3) LayerNorms
  ln_kernel<0><<<B_ * S_, 256, 0, stream>>>(q_x, qn_w, modF, qnB, S_);
  ln_kernel<1><<<B_ * K_, 256, 0, stream>>>(kv_x, kvn_w, modF, kvnB, K_);
  // 4) Q projection + RoPE
  gemm_wmma<EPI_ROPE_Q><<<mBlocks * (D_ / 64), 256, 0, stream>>>(
      qnB, WqT, M, D_, D_, cos_q, sin_q, nullptr, modF, qBf, nullptr, nullptr);
  // 5) KV projection (+RoPE on K, V stored transposed per head)
  gemm_wmma<EPI_KV><<<mBlocks * (2 * D_ / 64), 256, 0, stream>>>(
      kvnB, WkvT, M, 2 * D_, D_, cos_k, sin_k, nullptr, modF, kBf, nullptr, vTf);
  // 6) group-masked flash attention
  attn_kernel<<<(B_ * H_ * (S_ / 16)) / 8, 256, 0, stream>>>(qBf, kBf, vTf, gidx, attnB);
  // 7) Wo projection + residual + query gate
  gemm_wmma<EPI_WO><<<mBlocks * (D_ / 64), 256, 0, stream>>>(
      attnB, WoT, M, D_, D_, q_x, nullptr, qen, modF, nullptr, xF, nullptr);
  // 8) LN2 + mlp modulation (reuse qnB)
  ln_kernel<2><<<B_ * S_, 256, 0, stream>>>(xF, n2_w, modF, qnB, S_);
  // 9) MLP up + GELU
  gemm_wmma<EPI_GELU><<<mBlocks * (MR_ * D_ / 64), 256, 0, stream>>>(
      qnB, W1T, M, MR_ * D_, D_, b1, nullptr, nullptr, modF, h1B, nullptr, nullptr);
  // 10) MLP down + g_mlp + residual + gate -> output (fp32)
  gemm_wmma<EPI_FINAL><<<mBlocks * (D_ / 64), 256, 0, stream>>>(
      h1B, W2T, M, D_, MR_ * D_, b2, xF, qen, modF, nullptr, (float*)d_out, nullptr);
}
